// StandardAttention_26577257627869
// MI455X (gfx1250) — matrix operand order repaired and checked by reading, not compiled, not run on hardware
//
#include <hip/hip_runtime.h>
#include <math.h>
#include <stdint.h>

typedef __bf16 bf16;
typedef __bf16 bf16x4 __attribute__((ext_vector_type(4)));
typedef __bf16 bf16x8 __attribute__((ext_vector_type(8)));
typedef __bf16 v16bf  __attribute__((ext_vector_type(16)));
typedef float  v8f    __attribute__((ext_vector_type(8)));
typedef uint32_t u32;
typedef u32 u32x4 __attribute__((ext_vector_type(4)));
typedef u32 u32x8 __attribute__((ext_vector_type(8)));

#define DIM      1024
#define NHEADS   16
#define HEADDIM  64
#define SEQ      2048
#define BATCH    2
#define MROWS    (BATCH * SEQ)      // 4096

// ---------------- fragment loaders (ISA 7.12.2 layouts) ----------------
// A 16x32 bf16 fragment: lane = m + 16*g ; v0-3 <- K=g*8..+7 ; v4-7 <- K=16+g*8..+7
__device__ __forceinline__ v16bf load_frag_a(const bf16* lds, int stride, int m, int g, int koff) {
    union { v16bf v; bf16x8 h[2]; } u;
    u.h[0] = *(const bf16x8*)(lds + m * stride + koff + g * 8);
    u.h[1] = *(const bf16x8*)(lds + m * stride + koff + 16 + g * 8);
    return u.v;
}
// B 32x16 bf16 fragment from Bt[n][k] (K contiguous per column): lane n+16g holds K=g*8..+7, 16+g*8..+7
__device__ __forceinline__ v16bf load_frag_b(const bf16* ldsT, int stride, int n, int g, int koff) {
    union { v16bf v; bf16x8 h[2]; } u;
    u.h[0] = *(const bf16x8*)(ldsT + n * stride + koff + g * 8);
    u.h[1] = *(const bf16x8*)(ldsT + n * stride + koff + 16 + g * 8);
    return u.v;
}

#define WMMA_BF16(A, B, C) \
    __builtin_amdgcn_wmma_f32_16x16x32_bf16(false, (A), false, (B), (short)0, (C), false, false)

// ---------------- TDM: 2D tile DMA global -> LDS (D# per ISA 08 §8.3/8.4) ----------------
// dims/strides in elements (data_size = 2 bytes). pad codes: interval 0=2..7=256 dwords,
// amount 0=1..127=128 dwords. lds_off = byte offset in LDS.
__device__ __forceinline__ void tdm_load_2d(u32 lds_off, const void* gptr,
                                            u32 dim0, u32 dim1, uint64_t stride0,
                                            u32 tile0, u32 tile1,
                                            u32 pad_interval, u32 pad_amount) {
    uint64_t ga = (uint64_t)gptr;
    u32x4 g0;
    g0[0] = 1u;                                   // count=1 (valid user descriptor)
    g0[1] = lds_off;                              // lds_addr (bytes)
    g0[2] = (u32)ga;                              // global_addr[31:0]
    g0[3] = (u32)((ga >> 32) & 0x01FFFFFFu) | (2u << 30);  // addr[56:32] | type=2
    u32x8 g1;
    g1[0] = 0x00010000u                           // workgroup_mask=0, data_size=1 (2B)
          | (1u << 20)                            // pad_enable
          | (pad_interval << 22) | (pad_amount << 25);
    g1[1] = (dim0 & 0xFFFFu) << 16;               // tensor_dim0[15:0] at bits 63:48
    g1[2] = (dim0 >> 16) | ((dim1 & 0xFFFFu) << 16);
    g1[3] = (dim1 >> 16) | (tile0 << 16);         // tile_dim0
    g1[4] = tile1;                                // tile_dim1 (tile_dim2 = 0)
    g1[5] = (u32)stride0;                         // tensor_dim0_stride[31:0]
    g1[6] = (u32)(stride0 >> 32);                 // stride[47:32] | stride1[15:0]=0
    g1[7] = 0u;
    asm volatile("tensor_load_to_lds %0, %1" :: "s"(g0), "s"(g1) : "memory");
}

__device__ __forceinline__ u32 lds_off_u32(const void* p) {
    return (u32)(uintptr_t)p;                     // flat LDS addr: offset lives in low 32 bits
}

// ---------------- pre-conversion kernels ----------------
__global__ __launch_bounds__(256) void cvt_bf16(const float* __restrict__ in,
                                                bf16* __restrict__ out, int n4) {
    int i = blockIdx.x * 256 + threadIdx.x;
    if (i >= n4) return;
    float4 f = ((const float4*)in)[i];
    bf16x4 o;
    o[0] = (bf16)f.x; o[1] = (bf16)f.y; o[2] = (bf16)f.z; o[3] = (bf16)f.w;
    ((bf16x4*)out)[i] = o;
}

// in: f32 [R][C] -> out: bf16 [C][R]
__global__ __launch_bounds__(256) void transpose_cvt(const float* __restrict__ in,
                                                     bf16* __restrict__ out, int R, int C) {
    __shared__ float t[64][65];
    const int tid = threadIdx.x;
    const int c0 = blockIdx.x * 64, r0 = blockIdx.y * 64;
#pragma unroll
    for (int i = 0; i < 16; ++i) {
        int idx = tid + 256 * i;
        int r = idx >> 6, cc = idx & 63;
        t[r][cc] = in[(size_t)(r0 + r) * C + c0 + cc];
    }
    __syncthreads();
#pragma unroll
    for (int i = 0; i < 16; ++i) {
        int idx = tid + 256 * i;
        int rr = idx >> 6, cc = idx & 63;
        out[(size_t)(c0 + rr) * R + r0 + cc] = (bf16)t[cc][rr];
    }
}

// ---------------- TDM double-buffered WMMA GEMM ----------------
// A: bf16 [Mrows][Kdim] row-major. Bt: bf16 [Ncols][Kdim] (pre-transposed weights).
// EPI==0: qkv epilogue (bias + scatter q[B,H,N,D], k[B,H,N,D], vT[B,H,D,N], all bf16)
// EPI==1: proj epilogue (bias + f32 store to out[Mrows][Ncols])
template<int BM, int BN, int WROWS, int WCOLS, int EPI>
__global__ __launch_bounds__(32 * WROWS * WCOLS) void gemm_tdm(
    const bf16* __restrict__ A, const bf16* __restrict__ Bt,
    const float* __restrict__ bias, int Kdim, int Ncols, int Mrows,
    bf16* __restrict__ qd, bf16* __restrict__ kd, bf16* __restrict__ vd,
    float* __restrict__ out) {
    constexpr int LDT = 40;                       // padded row stride (32 + 8 bf16)
    __shared__ bf16 sA[2][BM * LDT];
    __shared__ bf16 sB[2][BN * LDT];
    const int tid = threadIdx.x, wave = tid >> 5, lane = tid & 31;
    const int wr = wave / WCOLS, wc = wave % WCOLS;
    const int g = lane >> 4, ln = lane & 15;
    const int m0 = blockIdx.x * BM, n0 = blockIdx.y * BN;
    constexpr int MF = BM / (WROWS * 16), NF = BN / (WCOLS * 16);

    v8f c[MF][NF];
#pragma unroll
    for (int i = 0; i < MF; ++i)
#pragma unroll
        for (int j = 0; j < NF; ++j) c[i][j] = (v8f)(0.0f);

    if (wave == 0) {   // prologue DMA (rows = 32 bf16 = 16 dwords -> interval code 3, pad 4 dwords -> code 3)
        tdm_load_2d(lds_off_u32(&sA[0][0]), A + (size_t)m0 * Kdim,
                    Kdim, Mrows - m0, Kdim, 32, BM, 3, 3);
        tdm_load_2d(lds_off_u32(&sB[0][0]), Bt + (size_t)n0 * Kdim,
                    Kdim, Ncols - n0, Kdim, 32, BN, 3, 3);
        __builtin_amdgcn_s_wait_tensorcnt(0);
    }
    __syncthreads();

    int buf = 0;
    for (int k0 = 0; k0 < Kdim; k0 += 32) {
        const int nk = k0 + 32;
        if (nk < Kdim && wave == 0) {
            tdm_load_2d(lds_off_u32(&sA[buf ^ 1][0]), A + (size_t)m0 * Kdim + nk,
                        Kdim - nk, Mrows - m0, Kdim, 32, BM, 3, 3);
            tdm_load_2d(lds_off_u32(&sB[buf ^ 1][0]), Bt + (size_t)n0 * Kdim + nk,
                        Kdim - nk, Ncols - n0, Kdim, 32, BN, 3, 3);
        }
        v16bf a[MF];
#pragma unroll
        for (int mi = 0; mi < MF; ++mi)
            a[mi] = load_frag_a(&sA[buf][0], LDT, (wr * MF + mi) * 16 + ln, g, 0);
#pragma unroll
        for (int j = 0; j < NF; ++j) {
            v16bf b = load_frag_b(&sB[buf][0], LDT, (wc * NF + j) * 16 + ln, g, 0);
#pragma unroll
            for (int mi = 0; mi < MF; ++mi) c[mi][j] = WMMA_BF16(a[mi], b, c[mi][j]);
        }
        if (wave == 0 && nk < Kdim) __builtin_amdgcn_s_wait_tensorcnt(0);
        __syncthreads();
        buf ^= 1;
    }

#pragma unroll
    for (int mi = 0; mi < MF; ++mi)
#pragma unroll
        for (int j = 0; j < NF; ++j)
#pragma unroll
            for (int r = 0; r < 8; ++r) {
                int row = m0 + (wr * MF + mi) * 16 + r + 8 * g;
                int col = n0 + (wc * NF + j) * 16 + ln;
                float val = c[mi][j][r] + bias[col];
                if (EPI == 0) {
                    int which = col >> 10;
                    int h = (col >> 6) & 15;
                    int d = col & 63;
                    int b = row >> 11;
                    int n = row & (SEQ - 1);
                    if (which == 0)
                        qd[(((size_t)b * NHEADS + h) * SEQ + n) * HEADDIM + d] = (bf16)val;
                    else if (which == 1)
                        kd[(((size_t)b * NHEADS + h) * SEQ + n) * HEADDIM + d] = (bf16)val;
                    else  // V stored transposed: [B,H,D,N]
                        vd[(((size_t)b * NHEADS + h) * HEADDIM + d) * SEQ + n] = (bf16)val;
                } else {
                    out[(size_t)row * DIM + col] = val;
                }
            }
}

// ---------------- flash attention, TDM double-buffered K / Vt tiles ----------------
__global__ __launch_bounds__(256) void attn_kernel(const bf16* __restrict__ qd,
                                                   const bf16* __restrict__ kd,
                                                   const bf16* __restrict__ vtd,  // [B,H,D,N]
                                                   bf16* __restrict__ od) {
    constexpr int LDT = 72;                       // 64 + 8 bf16 padded stride
    __shared__ bf16 sK[2][64 * LDT];              // [key][d]
    __shared__ bf16 sVt[2][64 * LDT];             // [d][key]
    __shared__ bf16 sP[8 * 16 * LDT];             // per-wave P scratch
    const int tid = threadIdx.x, wave = tid >> 5, lane = tid & 31;
    const int g = lane >> 4, ln = lane & 15;
    const int bh = blockIdx.y;
    const int q0 = blockIdx.x * 128;
    const bf16* qbh = qd + (size_t)bh * SEQ * HEADDIM;
    const bf16* kbh = kd + (size_t)bh * SEQ * HEADDIM;
    const bf16* vtbh = vtd + (size_t)bh * HEADDIM * SEQ;
    bf16* sp = sP + wave * 16 * LDT;
    const float scale = 0.125f;

    // per-wave Q A-fragments (16 rows x 64 d), straight from global bf16
    const int qrow = q0 + wave * 16 + ln;
    v16bf aq[2];
#pragma unroll
    for (int s = 0; s < 2; ++s) {
        union { v16bf v; bf16x8 h[2]; } u;
        u.h[0] = *(const bf16x8*)(qbh + (size_t)qrow * HEADDIM + s * 32 + g * 8);
        u.h[1] = *(const bf16x8*)(qbh + (size_t)qrow * HEADDIM + s * 32 + 16 + g * 8);
        aq[s] = u.v;
    }

    v8f o_acc[4];
#pragma unroll
    for (int j = 0; j < 4; ++j) o_acc[j] = (v8f)(0.0f);
    float m_i[8], l_i[8];
#pragma unroll
    for (int r = 0; r < 8; ++r) { m_i[r] = -INFINITY; l_i[r] = 0.0f; }

    // prologue DMA: rows = 64 bf16 = 32 dwords -> interval code 4, pad 4 dwords -> code 3
    if (wave == 0) {
        tdm_load_2d(lds_off_u32(&sK[0][0]), kbh, HEADDIM, SEQ, HEADDIM, 64, 64, 4, 3);
        tdm_load_2d(lds_off_u32(&sVt[0][0]), vtbh, SEQ, HEADDIM, SEQ, 64, 64, 4, 3);
        __builtin_amdgcn_s_wait_tensorcnt(0);
    }
    __syncthreads();

    int buf = 0;
    for (int k0 = 0; k0 < SEQ; k0 += 64) {
        const int nk0 = k0 + 64;
        if (nk0 < SEQ && wave == 0) {
            tdm_load_2d(lds_off_u32(&sK[buf ^ 1][0]), kbh + (size_t)nk0 * HEADDIM,
                        HEADDIM, SEQ - nk0, HEADDIM, 64, 64, 4, 3);
            tdm_load_2d(lds_off_u32(&sVt[buf ^ 1][0]), vtbh + nk0,
                        SEQ - nk0, HEADDIM, SEQ, 64, 64, 4, 3);
        }

        // S = scale * Q K^T
        v8f st[4];
#pragma unroll
        for (int j = 0; j < 4; ++j) {
            v8f acc = (v8f)(0.0f);
#pragma unroll
            for (int s = 0; s < 2; ++s) {
                v16bf b = load_frag_b(&sK[buf][0], LDT, j * 16 + ln, g, s * 32);
                acc = WMMA_BF16(aq[s], b, acc);
            }
#pragma unroll
            for (int r = 0; r < 8; ++r) acc[r] *= scale;
            st[j] = acc;
        }

        // online softmax (half-wave reductions stay within each 16-lane group)
        float alpha[8];
#pragma unroll
        for (int r = 0; r < 8; ++r) {
            float mx = st[0][r];
#pragma unroll
            for (int j = 1; j < 4; ++j) mx = fmaxf(mx, st[j][r]);
#pragma unroll
            for (int off = 8; off >= 1; off >>= 1) mx = fmaxf(mx, __shfl_xor(mx, off, 32));
            float mnew = fmaxf(m_i[r], mx);
            alpha[r] = __expf(m_i[r] - mnew);
            m_i[r] = mnew;
            float sum = 0.0f;
#pragma unroll
            for (int j = 0; j < 4; ++j) {
                float p = __expf(st[j][r] - mnew);
                st[j][r] = p;
                sum += p;
            }
#pragma unroll
            for (int off = 8; off >= 1; off >>= 1) sum += __shfl_xor(sum, off, 32);
            l_i[r] = l_i[r] * alpha[r] + sum;
        }

        // P -> per-wave LDS scratch -> A-fragments (DS in-order within wave)
#pragma unroll
        for (int j = 0; j < 4; ++j)
#pragma unroll
            for (int r = 0; r < 8; ++r)
                sp[(r + 8 * g) * LDT + j * 16 + ln] = (bf16)st[j][r];
        v16bf ap0 = load_frag_a(sp, LDT, ln, g, 0);
        v16bf ap1 = load_frag_a(sp, LDT, ln, g, 32);

        // O = O*alpha + P V
#pragma unroll
        for (int j = 0; j < 4; ++j) {
#pragma unroll
            for (int r = 0; r < 8; ++r) o_acc[j][r] *= alpha[r];
            v16bf b0 = load_frag_b(&sVt[buf][0], LDT, j * 16 + ln, g, 0);
            v16bf b1 = load_frag_b(&sVt[buf][0], LDT, j * 16 + ln, g, 32);
            o_acc[j] = WMMA_BF16(ap0, b0, o_acc[j]);
            o_acc[j] = WMMA_BF16(ap1, b1, o_acc[j]);
        }

        if (wave == 0 && nk0 < SEQ) __builtin_amdgcn_s_wait_tensorcnt(0);
        __syncthreads();
        buf ^= 1;
    }

    // finalize + store [B,N,C] bf16
    const int b = bh >> 4, h = bh & 15;
#pragma unroll
    for (int j = 0; j < 4; ++j)
#pragma unroll
        for (int r = 0; r < 8; ++r) {
            float val = o_acc[j][r] / l_i[r];
            int row = q0 + wave * 16 + r + 8 * g;
            int col = h * HEADDIM + j * 16 + ln;
            od[((size_t)b * SEQ + row) * DIM + col] = (bf16)val;
        }
}

// ---------------- launcher ----------------
extern "C" void kernel_launch(void* const* d_in, const int* in_sizes, int n_in,
                              void* d_out, int out_size, void* d_ws, size_t ws_size,
                              hipStream_t stream) {
    const float* x      = (const float*)d_in[0];
    const float* w_qkv  = (const float*)d_in[1];
    const float* b_qkv  = (const float*)d_in[2];
    const float* w_proj = (const float*)d_in[3];
    const float* b_proj = (const float*)d_in[4];
    float* out = (float*)d_out;

    const size_t per = (size_t)BATCH * NHEADS * SEQ * HEADDIM;   // 4M elems
    bf16* q    = (bf16*)d_ws;
    bf16* k    = q + per;
    bf16* vT   = k + per;                                        // [B,H,D,N]
    bf16* ao   = vT + per;                                       // [B,N,C]
    bf16* xb   = ao + per;                                       // [M,K] bf16
    bf16* wqT  = xb + (size_t)MROWS * DIM;                       // [3C,K]
    bf16* wpT  = wqT + (size_t)3 * DIM * DIM;                    // [C,K]

    cvt_bf16<<<(MROWS * DIM / 4 + 255) / 256, 256, 0, stream>>>(x, xb, MROWS * DIM / 4);
    transpose_cvt<<<dim3(3 * DIM / 64, DIM / 64), 256, 0, stream>>>(w_qkv, wqT, DIM, 3 * DIM);
    transpose_cvt<<<dim3(DIM / 64, DIM / 64), 256, 0, stream>>>(w_proj, wpT, DIM, DIM);

    gemm_tdm<128, 256, 2, 4, 0><<<dim3(MROWS / 128, 3 * DIM / 256), 256, 0, stream>>>(
        xb, wqT, b_qkv, DIM, 3 * DIM, MROWS, q, k, vT, nullptr);

    attn_kernel<<<dim3(SEQ / 128, BATCH * NHEADS), 256, 0, stream>>>(q, k, vT, ao);

    gemm_tdm<128, 128, 2, 2, 1><<<dim3(MROWS / 128, DIM / 128), 128, 0, stream>>>(
        ao, wpT, b_proj, DIM, DIM, MROWS, nullptr, nullptr, nullptr, out);
}
